// AttentionBlock_8538394984952
// MI455X (gfx1250) — compile-verified, matrix operand order checked by reading, not run on hardware
//
#include <hip/hip_runtime.h>

// ---------------------------------------------------------------------------
// AttentionBlock for MI455X (gfx1250): bf16 WMMA GEMMs + column-wise softmax
//   B=4, S=4096, D=256
//   Q/K/V = x @ W^T + b     (WMMA, bf16 in / f32 acc)
//   scores = Q @ K^T / 16   (WMMA, 64x32 wave tile, async-LDS-shared B tile)
//   attn   = softmax over QUERY axis (axis=1) -> column softmax, coalesced
//   out    = attn @ V       (WMMA, V stored transposed; async-LDS B tile)
// ---------------------------------------------------------------------------

typedef __attribute__((ext_vector_type(16))) __bf16 v16bf;
typedef __attribute__((ext_vector_type(8)))  __bf16 v8bf;
typedef __attribute__((ext_vector_type(8)))  float  v8f;

#define BATCH 4
#define SEQ   4096
#define DIM   256

// Load one 16x32 bf16 WMMA operand fragment from global memory.
// For A: base = row-major matrix, major0 = tile row.
// For B: base = column-major B (i.e. row-major B^T), major0 = tile col.
// Lane l (0-15): major index l, K-halves {0..7,16..23}; lanes 16-31: {8..15,24..31}.
static __device__ __forceinline__ v16bf load_frag(const __bf16* __restrict__ base,
                                                  int ld, int major0, int k0, int lane) {
    const int mj = major0 + (lane & 15);
    const int kb = k0 + ((lane >> 4) << 3);
    const __bf16* p = base + (size_t)mj * (size_t)ld + kb;
    union { v16bf v; v8bf h[2]; } u;
    u.h[0] = *(const v8bf*)(p);        // global_load_b128
    u.h[1] = *(const v8bf*)(p + 16);   // global_load_b128
    return u.v;
}

// Same fragment shape, read from an LDS-staged 32x32 tile (ld = 32).
static __device__ __forceinline__ v16bf lds_frag(const __bf16* base, int major0, int lane) {
    const int mj = major0 + (lane & 15);
    const int kb = (lane >> 4) << 3;
    const __bf16* p = base + mj * 32 + kb;
    union { v16bf v; v8bf h[2]; } u;
    u.h[0] = *(const v8bf*)(p);        // ds_load_b128
    u.h[1] = *(const v8bf*)(p + 16);   // ds_load_b128
    return u.v;
}

// Async DMA: 16 bytes global -> LDS per lane, tracked by ASYNCcnt.
// ISA 15.18.3 op 98: GLOBAL_LOAD_ASYNC_TO_LDS_B128 (VDST = LDS addr VGPR,
// VADDR = 64-bit global address pair, SADDR = off). Generic->LDS address is
// addr[31:0] per the flat-aperture rule, so a 32-bit truncation is the offset.
static __device__ __forceinline__ void async_lds_load_b128(const __bf16* gsrc,
                                                           __bf16* lds_dst) {
    const unsigned           lds_off = (unsigned)(uintptr_t)lds_dst;
    const unsigned long long gaddr   = (unsigned long long)(uintptr_t)gsrc;
    asm volatile("global_load_async_to_lds_b128 %0, %1, off"
                 :: "v"(lds_off), "v"(gaddr) : "memory");
}
static __device__ __forceinline__ void wait_asynccnt0() {
    asm volatile("s_wait_asynccnt 0x0" ::: "memory");
}

// ---------------------------------------------------------------------------
__global__ void cvt_f32_to_bf16(const float* __restrict__ src,
                                __bf16* __restrict__ dst, int n) {
    int i = blockIdx.x * blockDim.x + threadIdx.x;
    if (i < n) dst[i] = (__bf16)src[i];
}

// ---------------------------------------------------------------------------
// QKV projection: one wave per 16x16 output tile of one of {Q,K,V}.
// M dimension is flattened B*S. Row-major W serves as column-major B for
// y = x @ W^T. Writes Q,K bf16 row-major; V transposed (Vt[d][s]).
__global__ void __launch_bounds__(128)
qkv_proj_kernel(const __bf16* __restrict__ xh,
                const __bf16* __restrict__ Wqh, const __bf16* __restrict__ Wkh,
                const __bf16* __restrict__ Wvh,
                const float* __restrict__ bq, const float* __restrict__ bk,
                const float* __restrict__ bv,
                __bf16* __restrict__ Qh, __bf16* __restrict__ Kh,
                __bf16* __restrict__ Vt) {
    const int lane  = threadIdx.x & 31;
    const int wave  = (blockIdx.x << 2) + (threadIdx.x >> 5);
    const int mtile = wave / 48;          // (B*S)/16 = 1024 row tiles
    const int rest  = wave % 48;
    const int mat   = rest >> 4;          // 0=Q, 1=K, 2=V
    const int ntile = rest & 15;          // D/16 = 16 col tiles
    const int m0 = mtile * 16, n0 = ntile * 16;

    const __bf16* W   = (mat == 0) ? Wqh : (mat == 1) ? Wkh : Wvh;
    const float* bias = (mat == 0) ? bq  : (mat == 1) ? bk  : bv;

    v8f c = {};
    for (int k0 = 0; k0 < DIM; k0 += 32) {
        v16bf a = load_frag(xh, DIM, m0, k0, lane);
        v16bf b = load_frag(W,  DIM, n0, k0, lane);
        c = __builtin_amdgcn_wmma_f32_16x16x32_bf16(false, a, false, b,
                                                    (short)0, c, false, false);
    }

    const int   col  = n0 + (lane & 15);
    const int   roff = (lane >> 4) << 3;
    const float bv_  = bias[col];
    for (int r = 0; r < 8; ++r) {
        const int   row = m0 + roff + r;          // flattened b*S + s
        const float val = c[r] + bv_;
        if (mat == 0) {
            Qh[(size_t)row * DIM + col] = (__bf16)val;
        } else if (mat == 1) {
            Kh[(size_t)row * DIM + col] = (__bf16)val;
        } else {
            const int bb = row >> 12;             // row / SEQ
            const int s  = row & (SEQ - 1);       // row % SEQ
            Vt[((size_t)bb * DIM + col) * SEQ + s] = (__bf16)val;  // V^T
        }
    }
}

// ---------------------------------------------------------------------------
// scores[q,k] = (Q @ K^T)[q,k] / sqrt(D).
// Block = 4 waves stacked along M: block tile 256x32, wave tile 64x32
// (4 M-frags x 2 N-frags = 8 accumulators -> 8 WMMA per 6 fragment loads).
// The 32x32 B tile (rows of K = columns of B) is DMA'd once per block into
// LDS (global_load_async_to_lds_b128) and shared by all 4 waves.
__global__ void __launch_bounds__(128)
scores_kernel(const __bf16* __restrict__ Qh, const __bf16* __restrict__ Kh,
              float* __restrict__ scores, int batch) {
    __shared__ __bf16 Bs[32 * 32];                  // 2 KB of the 320 KB LDS

    const int lane = threadIdx.x & 31;
    const int wid  = threadIdx.x >> 5;
    const int bm   = blockIdx.x >> 7;               // / (SEQ/32 = 128)
    const int nt   = blockIdx.x & 127;
    const int m0   = bm * 256 + wid * 64;
    const int n0   = nt * 32;
    const __bf16* A  = Qh + (size_t)batch * SEQ * DIM;
    const __bf16* Bt = Kh + (size_t)batch * SEQ * DIM;

    const int crow = threadIdx.x >> 2;              // cooperative-load row 0..31
    const int cseg = (threadIdx.x & 3) << 3;        // 8-element segment

    v8f c[4][2] = {};
    for (int k0 = 0; k0 < DIM; k0 += 32) {
        __syncthreads();                            // prior tile fully consumed
        async_lds_load_b128(&Bt[(size_t)(n0 + crow) * DIM + k0 + cseg],
                            &Bs[crow * 32 + cseg]);
        wait_asynccnt0();                           // this wave's DMA landed
        __syncthreads();                            // all waves' segments visible

        v16bf b0 = lds_frag(Bs, 0,  lane);
        v16bf b1 = lds_frag(Bs, 16, lane);
        #pragma unroll
        for (int i = 0; i < 4; ++i) {
            v16bf a = load_frag(A, DIM, m0 + i * 16, k0, lane);
            c[i][0] = __builtin_amdgcn_wmma_f32_16x16x32_bf16(false, a, false, b0,
                                                              (short)0, c[i][0], false, false);
            c[i][1] = __builtin_amdgcn_wmma_f32_16x16x32_bf16(false, a, false, b1,
                                                              (short)0, c[i][1], false, false);
        }
    }

    const float scale = 0.0625f;                    // 1/sqrt(256)
    const int colb = lane & 15, roff = (lane >> 4) << 3;
    #pragma unroll
    for (int i = 0; i < 4; ++i)
        #pragma unroll
        for (int r = 0; r < 8; ++r) {
            const int row = m0 + i * 16 + roff + r;
            scores[(size_t)row * SEQ + (n0 + colb)]      = c[i][0][r] * scale;
            scores[(size_t)row * SEQ + (n0 + 16 + colb)] = c[i][1][r] * scale;
        }
}

// ---------------------------------------------------------------------------
// Column-wise softmax (softmax over axis=1 = query axis). Thread t owns
// column k: every access scores[q*S + k] is coalesced across the block.
// Two-pass online softmax; writes bf16 attn for the next WMMA GEMM.
__global__ void __launch_bounds__(256)
col_softmax_kernel(const float* __restrict__ scores, __bf16* __restrict__ attn) {
    const int k = blockIdx.x * blockDim.x + threadIdx.x;
    float m = -3.402823466e38f, s = 0.0f;
    for (int q = 0; q < SEQ; ++q) {
        const float v  = scores[(size_t)q * SEQ + k];
        const float nm = fmaxf(m, v);
        s = s * __expf(m - nm) + __expf(v - nm);    // v_exp_f32 (trans)
        m = nm;
    }
    const float inv = 1.0f / s;
    for (int q = 0; q < SEQ; ++q) {
        const float v = scores[(size_t)q * SEQ + k];
        attn[(size_t)q * SEQ + k] = (__bf16)(__expf(v - m) * inv);
    }
}

// ---------------------------------------------------------------------------
// out[q,d] = sum_k attn[q,k] * V[k,d]. B operand needs column-major V, which
// is exactly the row-major V^T stored by the projection epilogue.
// Same blocking as scores_kernel: 256x32 block tile, 64x32 wave tile,
// async-LDS-shared 32x32 B tile, K = 4096.
__global__ void __launch_bounds__(128)
attn_v_kernel(const __bf16* __restrict__ attn, const __bf16* __restrict__ Vt,
              float* __restrict__ out, int batch) {
    __shared__ __bf16 Bs[32 * 32];

    const int lane = threadIdx.x & 31;
    const int wid  = threadIdx.x >> 5;
    const int bm   = blockIdx.x >> 3;               // / (DIM/32 = 8)
    const int nt   = blockIdx.x & 7;
    const int m0   = bm * 256 + wid * 64;
    const int n0   = nt * 32;
    const __bf16* A = attn;                         // per-batch buffer [S x S]
    const __bf16* B = Vt + (size_t)batch * DIM * SEQ;

    const int crow = threadIdx.x >> 2;
    const int cseg = (threadIdx.x & 3) << 3;

    v8f c[4][2] = {};
    for (int k0 = 0; k0 < SEQ; k0 += 32) {
        __syncthreads();
        async_lds_load_b128(&B[(size_t)(n0 + crow) * SEQ + k0 + cseg],
                            &Bs[crow * 32 + cseg]);
        wait_asynccnt0();
        __syncthreads();

        if (k0 + 128 < SEQ)                         // -> global_prefetch_b8
            __builtin_prefetch(A + (size_t)(m0 + (lane & 15)) * SEQ + k0 + 128, 0, 0);

        v16bf b0 = lds_frag(Bs, 0,  lane);
        v16bf b1 = lds_frag(Bs, 16, lane);
        #pragma unroll
        for (int i = 0; i < 4; ++i) {
            v16bf a = load_frag(A, SEQ, m0 + i * 16, k0, lane);
            c[i][0] = __builtin_amdgcn_wmma_f32_16x16x32_bf16(false, a, false, b0,
                                                              (short)0, c[i][0], false, false);
            c[i][1] = __builtin_amdgcn_wmma_f32_16x16x32_bf16(false, a, false, b1,
                                                              (short)0, c[i][1], false, false);
        }
    }

    float* O = out + (size_t)batch * SEQ * DIM;
    const int colb = lane & 15, roff = (lane >> 4) << 3;
    #pragma unroll
    for (int i = 0; i < 4; ++i)
        #pragma unroll
        for (int r = 0; r < 8; ++r) {
            const int row = m0 + i * 16 + roff + r;
            O[(size_t)row * DIM + (n0 + colb)]      = c[i][0][r];
            O[(size_t)row * DIM + (n0 + 16 + colb)] = c[i][1][r];
        }
}

// ---------------------------------------------------------------------------
extern "C" void kernel_launch(void* const* d_in, const int* in_sizes, int n_in,
                              void* d_out, int out_size, void* d_ws, size_t ws_size,
                              hipStream_t stream) {
    (void)in_sizes; (void)n_in; (void)out_size; (void)ws_size;
    const float* x  = (const float*)d_in[0];
    const float* Wq = (const float*)d_in[1];
    const float* bq = (const float*)d_in[2];
    const float* Wk = (const float*)d_in[3];
    const float* bk = (const float*)d_in[4];
    const float* Wv = (const float*)d_in[5];
    const float* bv = (const float*)d_in[6];
    float* out = (float*)d_out;

    char* ws = (char*)d_ws;
    size_t off = 0;
    auto alloc = [&](size_t bytes) -> void* {
        void* p = ws + off;
        off += (bytes + 255) & ~(size_t)255;
        return p;
    };
    const size_t NX = (size_t)BATCH * SEQ * DIM;
    __bf16* xh     = (__bf16*)alloc(NX * 2);
    __bf16* Wqh    = (__bf16*)alloc((size_t)DIM * DIM * 2);
    __bf16* Wkh    = (__bf16*)alloc((size_t)DIM * DIM * 2);
    __bf16* Wvh    = (__bf16*)alloc((size_t)DIM * DIM * 2);
    __bf16* Qh     = (__bf16*)alloc(NX * 2);
    __bf16* Kh     = (__bf16*)alloc(NX * 2);
    __bf16* Vt     = (__bf16*)alloc(NX * 2);
    float*  scores = (float*)alloc((size_t)SEQ * SEQ * 4);   // reused per batch
    __bf16* attn   = (__bf16*)alloc((size_t)SEQ * SEQ * 2);  // reused per batch

    // 1) f32 -> bf16 conversions
    cvt_f32_to_bf16<<<(int)((NX + 255) / 256), 256, 0, stream>>>(x, xh, (int)NX);
    cvt_f32_to_bf16<<<(DIM * DIM + 255) / 256, 256, 0, stream>>>(Wq, Wqh, DIM * DIM);
    cvt_f32_to_bf16<<<(DIM * DIM + 255) / 256, 256, 0, stream>>>(Wk, Wkh, DIM * DIM);
    cvt_f32_to_bf16<<<(DIM * DIM + 255) / 256, 256, 0, stream>>>(Wv, Wvh, DIM * DIM);

    // 2) QKV projection: (B*S/16) * 3 * (D/16) tiles, 4 waves/block
    {
        const int waves = (BATCH * SEQ / 16) * 3 * (DIM / 16);   // 49152
        qkv_proj_kernel<<<waves / 4, 128, 0, stream>>>(xh, Wqh, Wkh, Wvh,
                                                       bq, bk, bv, Qh, Kh, Vt);
    }

    // 3) Per-batch: scores -> column softmax -> attn @ V (scratch reused,
    //    safe because all launches are ordered on the same stream)
    for (int b = 0; b < BATCH; ++b) {
        const int sblocks = (SEQ / 256) * (SEQ / 32);            // 2048
        scores_kernel<<<sblocks, 128, 0, stream>>>(Qh, Kh, scores, b);
        col_softmax_kernel<<<SEQ / 256, 256, 0, stream>>>(scores, attn);
        const int oblocks = (SEQ / 256) * (DIM / 32);            // 128
        attn_v_kernel<<<oblocks, 128, 0, stream>>>(attn, Vt, out, b);
    }
}
